// HDMR_Network_68015102099709
// MI455X (gfx1250) — compile-verified, matrix operand order checked by reading, not run on hardware
//
#include <hip/hip_runtime.h>
#include <hip/hip_bf16.h>
#include <stdint.h>

typedef __attribute__((ext_vector_type(16))) _Float16 v16h;
typedef __attribute__((ext_vector_type(8)))  float    v8f;

#define NVAR 8
#define HID  128
#define NL   2
#define BATCH 8192
#define NGRP 92                 // 8 singles + 28 pairs + 56 triples
#define ROWS_PER_WAVE 32        // two 16-row M-tiles per wave: B reuse x2
#define WAVES_PER_BLOCK 4
#define ROWS_PER_BLOCK (ROWS_PER_WAVE*WAVES_PER_BLOCK)   // 128
#define NBTILES (BATCH/ROWS_PER_BLOCK)                   // 64
#define FRAG_HALVES 512         // 32 lanes * 16 halves (one 16x32 B fragment)
#define FRAGS_PER_LAYER 32      // 8 n-tiles * 4 k-steps
#define WSH_HALVES ((size_t)NGRP*NL*FRAGS_PER_LAYER*FRAG_HALVES)  // 3,014,656 halves

// Variable membership per group (itertools.combinations lexicographic order).
__constant__ int GV[NGRP][3] = {
  // singles
  {0,0,0},{1,0,0},{2,0,0},{3,0,0},{4,0,0},{5,0,0},{6,0,0},{7,0,0},
  // pairs (28)
  {0,1,0},{0,2,0},{0,3,0},{0,4,0},{0,5,0},{0,6,0},{0,7,0},
  {1,2,0},{1,3,0},{1,4,0},{1,5,0},{1,6,0},{1,7,0},
  {2,3,0},{2,4,0},{2,5,0},{2,6,0},{2,7,0},
  {3,4,0},{3,5,0},{3,6,0},{3,7,0},
  {4,5,0},{4,6,0},{4,7,0},
  {5,6,0},{5,7,0},
  {6,7,0},
  // triples (56)
  {0,1,2},{0,1,3},{0,1,4},{0,1,5},{0,1,6},{0,1,7},
  {0,2,3},{0,2,4},{0,2,5},{0,2,6},{0,2,7},
  {0,3,4},{0,3,5},{0,3,6},{0,3,7},
  {0,4,5},{0,4,6},{0,4,7},
  {0,5,6},{0,5,7},
  {0,6,7},
  {1,2,3},{1,2,4},{1,2,5},{1,2,6},{1,2,7},
  {1,3,4},{1,3,5},{1,3,6},{1,3,7},
  {1,4,5},{1,4,6},{1,4,7},
  {1,5,6},{1,5,7},
  {1,6,7},
  {2,3,4},{2,3,5},{2,3,6},{2,3,7},
  {2,4,5},{2,4,6},{2,4,7},
  {2,5,6},{2,5,7},
  {2,6,7},
  {3,4,5},{3,4,6},{3,4,7},
  {3,5,6},{3,5,7},
  {3,6,7},
  {4,5,6},{4,5,7},
  {4,6,7},
  {5,6,7}
};

// Telescoped HDMR weight of each group's raw MLP output (derivation in round 0;
// includes the reference's f_jj[var-index] indexing quirk).
__device__ __forceinline__ float group_coef(int g) {
  if (g == 0)  return 120.0f;   // var 0
  if (g <= 2)  return 15.0f;    // vars 1,2
  if (g < 8)   return -6.0f;    // vars 3..7
  if (g < 16)  return -20.0f;   // pair rows 0..7
  return 1.0f;                  // remaining pairs + all triples
}

__device__ __forceinline__ float fast_tanh(float h) {
#if __has_builtin(__builtin_amdgcn_tanhf)
  return __builtin_amdgcn_tanhf(h);
#elif __has_builtin(__builtin_amdgcn_tanh_f32)
  return __builtin_amdgcn_tanh_f32(h);
#else
  float e = __expf(-2.0f * h);
  return (1.0f - e) / (1.0f + e);
#endif
}

// sigmoid(2h) = 0.5 + 0.5*tanh(h); outer fma + f16 cvt fold into v_fma_mixlo_f16.
__device__ __forceinline__ _Float16 act_from_tanh(float t) {
  return (_Float16)__builtin_fmaf(0.5f, t, 0.5f);
}

// k-position within a 16-bit A/B operand fragment for (vgpr pair p, lane-half hi):
// VGPR0..3 -> K 0..7 (lanes 0-15) / K 8..15 (lanes 16-31); VGPR4..7 -> K 16..23 / 24..31.
__device__ __host__ __forceinline__ int frag_kpair(int p, int hi) {
  return ((p < 4) ? 2 * p : 16 + 2 * (p - 4)) + (hi << 3);
}

// ---------------------------------------------------------------------------
// Kernel 1: f32 -> f16 hidden weights, pre-swizzled into B-fragment order.
// Dest half index == tid: (((g*2+l)*32 + nt*4 + ks)*512 + lane*16 + h).
// ---------------------------------------------------------------------------
__global__ __launch_bounds__(256) void hdmr_swizzle(
    const float* __restrict__ Wh1, const float* __restrict__ Wh2,
    const float* __restrict__ Wh3, _Float16* __restrict__ wsH) {
  size_t tid = (size_t)blockIdx.x * blockDim.x + threadIdx.x;
  if (tid >= WSH_HALVES) return;
  int h  = (int)(tid & 15);
  int L  = (int)((tid >> 4) & 31);
  int ks = (int)((tid >> 9) & 3);
  int nt = (int)((tid >> 11) & 7);
  int l  = (int)((tid >> 14) & 1);
  int g  = (int)(tid >> 15);
  const float* W; int n;
  if (g < 8)       { W = Wh1; n = g; }
  else if (g < 36) { W = Wh2; n = g - 8; }
  else             { W = Wh3; n = g - 36; }
  int ncol = nt * 16 + (L & 15);                       // output channel (B column)
  int v = h >> 1;
  int k = ks * 32 + frag_kpair(v, (L >= 16) ? 1 : 0) + (h & 1);  // input channel
  // B[k][ncol] = W[ncol][k];  W_h layout: [n][NL][HID(out)][HID(in)]
  float val = W[(((size_t)n * NL + l) * HID + ncol) * HID + k];
  wsH[tid] = (_Float16)val;
}

// ---------------------------------------------------------------------------
// Kernel 2: grouped MLP. grid = (64 batch tiles, 92 groups), 128 threads.
// Each wave32 owns 32 batch rows (two 16-row M-tiles) end-to-end; every B
// fragment load feeds two WMMAs. No inter-wave sync needed.
// ---------------------------------------------------------------------------
__global__ __launch_bounds__(128) void hdmr_mlp(
    const float* __restrict__ x,
    const float* __restrict__ Wi1, const float* __restrict__ bi1,
    const float* __restrict__ bh1, const float* __restrict__ Wo1,
    const float* __restrict__ bo1,
    const float* __restrict__ Wi2, const float* __restrict__ bi2,
    const float* __restrict__ bh2, const float* __restrict__ Wo2,
    const float* __restrict__ bo2,
    const float* __restrict__ Wi3, const float* __restrict__ bi3,
    const float* __restrict__ bh3, const float* __restrict__ Wo3,
    const float* __restrict__ bo3,
    const _Float16* __restrict__ wsH, float* __restrict__ wsY) {
  __shared__ _Float16 sAct[WAVES_PER_BLOCK][ROWS_PER_WAVE * HID];  // 32 KB
  __shared__ float    sX[WAVES_PER_BLOCK][ROWS_PER_WAVE][4];       // 2 KB

  const int g    = blockIdx.y;
  const int wid  = threadIdx.x >> 5;
  const int lane = threadIdx.x & 31;
  const int row  = lane & 15;
  const int hi   = lane >> 4;
  const int rowBase = blockIdx.x * ROWS_PER_BLOCK + wid * ROWS_PER_WAVE;

  // Resolve this group's parameter slices.
  int n, d;
  const float *Wi, *bi, *bh, *Wo, *boP;
  if (g < 8)       { n = g;      d = 1; Wi = Wi1; bi = bi1; bh = bh1; Wo = Wo1; boP = bo1; }
  else if (g < 36) { n = g - 8;  d = 2; Wi = Wi2; bi = bi2; bh = bh2; Wo = Wo2; boP = bo2; }
  else             { n = g - 36; d = 3; Wi = Wi3; bi = bi3; bh = bh3; Wo = Wo3; boP = bo3; }
  Wi += (size_t)n * HID * d;
  bi += (size_t)n * HID;
  bh += (size_t)n * NL * HID;
  Wo += (size_t)n * HID;
  const float bo = boP[n];

  // Gather this wave's x columns into LDS (one batch row per lane).
  #pragma unroll
  for (int i = 0; i < 3; ++i)
    sX[wid][lane][i] = (i < d) ? x[(size_t)(rowBase + lane) * NVAR + GV[g][i]] : 0.0f;

  // ---- Input layer (K = d <= 3: VALU, not worth WMMA) ------------------------
  // Pre-halved weights/bias -> pure FMA pre-activation; activations batched in
  // phases of 8 so the v_tanh_f32 ops pipeline without TRANS-dep stalls.
  {
    #pragma unroll
    for (int cc = 0; cc < 4; ++cc) {
      int c = lane * 4 + cc;
      float hw0 = 0.5f * Wi[c * d + 0];
      float hw1 = (d > 1) ? 0.5f * Wi[c * d + 1] : 0.0f;
      float hw2 = (d > 2) ? 0.5f * Wi[c * d + 2] : 0.0f;
      float hb  = 0.5f * bi[c];
      for (int rb = 0; rb < ROWS_PER_WAVE; rb += 8) {
        float hv[8];
        #pragma unroll
        for (int j = 0; j < 8; ++j) {
          int r = rb + j;
          hv[j] = __builtin_fmaf(hw0, sX[wid][r][0],
                  __builtin_fmaf(hw1, sX[wid][r][1],
                  __builtin_fmaf(hw2, sX[wid][r][2], hb)));
        }
        #pragma unroll
        for (int j = 0; j < 8; ++j) hv[j] = fast_tanh(hv[j]);
        #pragma unroll
        for (int j = 0; j < 8; ++j)
          sAct[wid][(rb + j) * HID + c] = act_from_tanh(hv[j]);
      }
    }
  }

  // ---- Hidden layers: 2 x (8 n-tiles x 4 k-steps x 2 M-tiles) WMMAs ---------
  union AFrag { uint32_t u[8]; v16h v; };
  for (int l = 0; l < NL; ++l) {
    // Build A fragments (16x32 f16, documented CDNA5 layout) for both M-tiles.
    AFrag a0[4], a1[4];
    #pragma unroll
    for (int ks = 0; ks < 4; ++ks) {
      #pragma unroll
      for (int p = 0; p < 8; ++p) {
        int kb = frag_kpair(p, hi);          // even -> coalesces into ds_load_b128
        int col = ks * 32 + kb;
        a0[ks].u[p] = *(const uint32_t*)&sAct[wid][row * HID + col];
        a1[ks].u[p] = *(const uint32_t*)&sAct[wid][(row + 16) * HID + col];
      }
    }
    const _Float16* fragBase =
        wsH + ((size_t)(g * NL + l) * FRAGS_PER_LAYER) * FRAG_HALVES;
    #pragma unroll
    for (int nt = 0; nt < 8; ++nt) {
      v8f c0 = {};   // zero accumulators -> inline-0 SRC2 on first WMMA
      v8f c1 = {};
      #pragma unroll
      for (int ks = 0; ks < 4; ++ks) {
        v16h b = *(const v16h*)(fragBase +
                                (size_t)(nt * 4 + ks) * FRAG_HALVES + lane * 16);
        c0 = __builtin_amdgcn_wmma_f32_16x16x32_f16(
            false, a0[ks].v, false, b, (short)0, c0, false, false);
        c1 = __builtin_amdgcn_wmma_f32_16x16x32_f16(
            false, a1[ks].v, false, b, (short)0, c1, false, false);
      }
      // Batched epilogue: (1) fma pre-act, (2) back-to-back tanh, (3) cvt+store.
      float hb = 0.5f * bh[l * HID + nt * 16 + row];  // bias for D col N = lane%16
      float hv[16];
      #pragma unroll
      for (int r = 0; r < 8; ++r) {
        hv[r]     = __builtin_fmaf(c0[r], 0.5f, hb);
        hv[r + 8] = __builtin_fmaf(c1[r], 0.5f, hb);
      }
      #pragma unroll
      for (int r = 0; r < 16; ++r) hv[r] = fast_tanh(hv[r]);
      // D layout: VGPR r -> row M=r (lanes 0-15) / M=r+8 (lanes 16-31), col = lane%16
      #pragma unroll
      for (int r = 0; r < 8; ++r) {
        sAct[wid][(r + hi * 8) * HID + nt * 16 + row]      = act_from_tanh(hv[r]);
        sAct[wid][(r + hi * 8 + 16) * HID + nt * 16 + row] = act_from_tanh(hv[r + 8]);
      }
    }
  }

  // ---- Output layer: scalar head, split K across lane halves, shfl-combine ---
  float sum0 = 0.0f, sum1 = 0.0f;
  const int h0 = hi * 64;
  #pragma unroll 8
  for (int h = 0; h < 64; ++h) {
    float w = Wo[h0 + h];
    sum0 += (float)sAct[wid][row * HID + h0 + h] * w;
    sum1 += (float)sAct[wid][(row + 16) * HID + h0 + h] * w;
  }
  sum0 += __shfl_xor(sum0, 16, 32);
  sum1 += __shfl_xor(sum1, 16, 32);
  if (lane < 16) {
    float cf = group_coef(g);
    wsY[(size_t)g * BATCH + rowBase + row]      = cf * (sum0 + bo);
    wsY[(size_t)g * BATCH + rowBase + row + 16] = cf * (sum1 + bo);
  }
}

// ---------------------------------------------------------------------------
// Kernel 3: deterministic reduction over groups (no float atomics).
// ---------------------------------------------------------------------------
__global__ __launch_bounds__(256) void hdmr_reduce(
    const float* __restrict__ wsY, const float* __restrict__ f0,
    float* __restrict__ out) {
  int b = blockIdx.x * blockDim.x + threadIdx.x;
  if (b >= BATCH) return;
  float s = -35.0f * f0[0];
  for (int g = 0; g < NGRP; ++g) s += wsY[(size_t)g * BATCH + b];
  out[b] = s;
}

extern "C" void kernel_launch(void* const* d_in, const int* in_sizes, int n_in,
                              void* d_out, int out_size, void* d_ws, size_t ws_size,
                              hipStream_t stream) {
  (void)in_sizes; (void)n_in; (void)out_size; (void)ws_size;
  const float* x   = (const float*)d_in[0];
  const float* f0  = (const float*)d_in[1];
  const float* Wi1 = (const float*)d_in[2],  *bi1 = (const float*)d_in[3];
  const float* Wh1 = (const float*)d_in[4],  *bh1 = (const float*)d_in[5];
  const float* Wo1 = (const float*)d_in[6],  *bo1 = (const float*)d_in[7];
  const float* Wi2 = (const float*)d_in[8],  *bi2 = (const float*)d_in[9];
  const float* Wh2 = (const float*)d_in[10], *bh2 = (const float*)d_in[11];
  const float* Wo2 = (const float*)d_in[12], *bo2 = (const float*)d_in[13];
  const float* Wi3 = (const float*)d_in[14], *bi3 = (const float*)d_in[15];
  const float* Wh3 = (const float*)d_in[16], *bh3 = (const float*)d_in[17];
  const float* Wo3 = (const float*)d_in[18], *bo3 = (const float*)d_in[19];

  _Float16* wsH = (_Float16*)d_ws;                                   // ~6.0 MB
  size_t yOff = ((WSH_HALVES * sizeof(_Float16)) + 255) & ~(size_t)255;
  float* wsY = (float*)((char*)d_ws + yOff);                         // ~3.0 MB

  int swzBlocks = (int)((WSH_HALVES + 255) / 256);
  hdmr_swizzle<<<swzBlocks, 256, 0, stream>>>(Wh1, Wh2, Wh3, wsH);

  dim3 grid(NBTILES, NGRP);   // 64 x 92 = 5888 workgroups of 4 waves
  hdmr_mlp<<<grid, 128, 0, stream>>>(x,
      Wi1, bi1, bh1, Wo1, bo1,
      Wi2, bi2, bh2, Wo2, bo2,
      Wi3, bi3, bh3, Wo3, bo3,
      wsH, wsY);

  hdmr_reduce<<<(BATCH + 255) / 256, 256, 0, stream>>>(wsY, f0, (float*)d_out);
}
